// GlobalBlock_52432960749580
// MI455X (gfx1250) — compile-verified, matrix operand order checked by reading, not run on hardware
//
#include <hip/hip_runtime.h>
#include <hip/hip_bf16.h>
#include <math.h>

// ---------------------------------------------------------------- types
typedef _Float16 h16;
typedef __attribute__((ext_vector_type(16))) _Float16 v16h;
typedef __attribute__((ext_vector_type(8)))  _Float16 v8h;
typedef __attribute__((ext_vector_type(8)))  float    v8f;

// ---------------------------------------------------------------- problem dims
#define B_   16
#define N_   1024
#define C_   1024
#define H_   16
#define HD_  64
#define R_   128
#define N2_  512          // N/2
#define NU_  384          // N2 - R (unmerged)
#define NM_  896          // N2 + NU (tokens after merge)
#define C4_  4096
#define M1_  (B_*N_)      // 16384 tokens pre-merge
#define M2_  (B_*NM_)     // 14336 tokens post-merge

// ---------------------------------------------------------------- helpers
__device__ __forceinline__ v8f wmma_f16(v16h a, v16h b, v8f c) {
    return __builtin_amdgcn_wmma_f32_16x16x32_f16(
        false, a, false, b, (short)0, c, false, false);
}

// Load a 16x32 f16 A/B fragment (row-major source, rows of length `ld`).
// Per CDNA5 16-bit layout: lane&15 selects row; lanes>=16 take K offset +8;
// vgprs 0-3 hold K base+[0..7], vgprs 4-7 hold K base+16+[0..7].
// => two contiguous 16-byte loads per lane.
__device__ __forceinline__ v16h load_frag(const h16* __restrict__ base,
                                          int ld, int row, int kbase, int lane) {
    const h16* p = base + (size_t)row * ld + kbase + ((lane >> 4) << 3);
    v8h lo = *(const v8h*)(p);
    v8h hi = *(const v8h*)(p + 16);
    v16h r;
#pragma unroll
    for (int i = 0; i < 8; ++i) { r[i] = lo[i]; r[i + 8] = hi[i]; }
    return r;
}

__device__ __forceinline__ float redmax16(float v) {
    v = fmaxf(v, __shfl_xor(v, 1, 32));
    v = fmaxf(v, __shfl_xor(v, 2, 32));
    v = fmaxf(v, __shfl_xor(v, 4, 32));
    v = fmaxf(v, __shfl_xor(v, 8, 32));
    return v;
}
__device__ __forceinline__ float redsum16(float v) {
    v += __shfl_xor(v, 1, 32);
    v += __shfl_xor(v, 2, 32);
    v += __shfl_xor(v, 4, 32);
    v += __shfl_xor(v, 8, 32);
    return v;
}

__device__ __forceinline__ float gelu_exact(float v) {
    return 0.5f * v * (1.0f + erff(v * 0.7071067811865475f));
}

// ---------------------------------------------------------------- tiny kernels
__global__ __launch_bounds__(256) void cvt_f32_to_f16(const float* __restrict__ s,
                                                      h16* __restrict__ d, int n) {
    int i = blockIdx.x * 1024 + threadIdx.x;
#pragma unroll
    for (int k = 0; k < 4; ++k) {
        int idx = i + k * 256;
        if (idx < n) d[idx] = (h16)s[idx];
    }
}

__global__ __launch_bounds__(256) void log_kernel(const float* __restrict__ s,
                                                  float* __restrict__ d, int n) {
    int i = blockIdx.x * 256 + threadIdx.x;
    if (i < n) d[i] = logf(s[i]);
}

// LayerNorm (C=1024), f32 in -> f16 out
__global__ __launch_bounds__(256) void ln_kernel(const float* __restrict__ x,
                                                 const float* __restrict__ gw,
                                                 const float* __restrict__ bw,
                                                 h16* __restrict__ out) {
    int row = blockIdx.x;
    const float* xr = x + (size_t)row * C_;
    float s = 0.f, s2 = 0.f;
    for (int c = threadIdx.x; c < C_; c += 256) {
        float v = xr[c]; s += v; s2 += v * v;
    }
    for (int m = 1; m < 32; m <<= 1) { s += __shfl_xor(s, m, 32); s2 += __shfl_xor(s2, m, 32); }
    __shared__ float sh1[8], sh2[8];
    int lane = threadIdx.x & 31, w = threadIdx.x >> 5;
    if (lane == 0) { sh1[w] = s; sh2[w] = s2; }
    __syncthreads();
    s = 0.f; s2 = 0.f;
#pragma unroll
    for (int i = 0; i < 8; ++i) { s += sh1[i]; s2 += sh2[i]; }
    float mean = s * (1.f / C_);
    float var  = s2 * (1.f / C_) - mean * mean;
    float rstd = rsqrtf(var + 1e-5f);
    h16* orow = out + (size_t)row * C_;
    for (int c = threadIdx.x; c < C_; c += 256)
        orow[c] = (h16)((xr[c] - mean) * rstd * gw[c] + bw[c]);
}

// ---------------------------------------------------------------- WMMA GEMM
// out[M,Nc] = act( A[M,K](f16) @ W[Nc,K]^T (f16) + bias[Nc] ) (+ resid[M,Nc])
// ACT: 0=none, 1=gelu.  Block = 256 thr (8 waves), tile 64x64, wave does 16x32.
template <int ACT, int RES, int OUT16>
__global__ __launch_bounds__(256) void gemm_wmma(const h16* __restrict__ A,
                                                 const h16* __restrict__ W,
                                                 const float* __restrict__ bias,
                                                 const float* __restrict__ resid,
                                                 float* __restrict__ outF,
                                                 h16* __restrict__ outH,
                                                 int M, int Nc, int K) {
    int lane = threadIdx.x & 31, w = threadIdx.x >> 5;
    int tilesN = Nc >> 6;
    int bm = blockIdx.x / tilesN, bn = blockIdx.x % tilesN;
    int row0 = bm * 64 + (w & 3) * 16;
    int col0 = bn * 64 + (w >> 2) * 32;
    (void)M;
    v8f acc0 = {}; v8f acc1 = {};
    int arow  = row0 + (lane & 15);
    int brow0 = col0 + (lane & 15);
    int brow1 = brow0 + 16;
    for (int k = 0; k < K; k += 32) {
        __builtin_prefetch(A + (size_t)arow  * K + k + 128, 0, 1);
        __builtin_prefetch(W + (size_t)brow0 * K + k + 128, 0, 1);
        v16h af  = load_frag(A, K, arow,  k, lane);
        v16h bf0 = load_frag(W, K, brow0, k, lane);
        v16h bf1 = load_frag(W, K, brow1, k, lane);
        acc0 = wmma_f16(af, bf0, acc0);
        acc1 = wmma_f16(af, bf1, acc1);
    }
    int mbase = row0 + ((lane >> 4) << 3);
    int n0 = col0 + (lane & 15);
    float b0 = bias[n0], b1 = bias[n0 + 16];
#pragma unroll
    for (int g = 0; g < 8; ++g) {
        int m = mbase + g;
        float v0 = acc0[g] + b0;
        float v1 = acc1[g] + b1;
        if (ACT == 1) { v0 = gelu_exact(v0); v1 = gelu_exact(v1); }
        size_t o0 = (size_t)m * Nc + n0;
        if (RES) { v0 += resid[o0]; v1 += resid[o0 + 16]; }
        if (OUT16) { outH[o0] = (h16)v0; outH[o0 + 16] = (h16)v1; }
        else       { outF[o0] = v0;      outF[o0 + 16] = v1; }
    }
}

// ---------------------------------------------------------------- V transpose
// vt[(b*H+h)][d][n] = v[b, n, h*HD+d]  (per-head 64 x 1024, keys contiguous)
// LDS tile transpose; all global traffic is 16-byte vectors.
__global__ __launch_bounds__(256) void vtrans_kernel(const h16* __restrict__ qkv,
                                                     h16* __restrict__ vt) {
    __shared__ h16 tile[64][72];              // [d][key], pitch 144B (16B-aligned)
    int bh = blockIdx.x >> 4;                 // 0..255  (b*16+h)
    int nt = blockIdx.x & 15;                 // 64-key tile
    int b = bh >> 4, h = bh & 15;
    int n0 = nt * 64;
    const h16* vp = qkv + (size_t)b * N_ * (3 * C_) + 2 * C_ + h * HD_;
#pragma unroll
    for (int it = 0; it < 2; ++it) {
        int idx = threadIdx.x + it * 256;     // 0..511
        int key = idx >> 3;                   // 0..63
        int d8  = (idx & 7) * 8;
        v8h val = *(const v8h*)(vp + (size_t)(n0 + key) * (3 * C_) + d8);
#pragma unroll
        for (int e = 0; e < 8; ++e) tile[d8 + e][key] = val[e];
    }
    __syncthreads();
    h16* op = vt + ((size_t)bh * HD_) * N_ + n0;
#pragma unroll
    for (int it = 0; it < 2; ++it) {
        int idx = threadIdx.x + it * 256;
        int d  = idx >> 3;
        int k8 = (idx & 7) * 8;
        *(v8h*)(op + (size_t)d * N_ + k8) = *(const v8h*)(&tile[d][k8]);
    }
}

// ---------------------------------------------------------------- attention
// One wave = (b, h, 16 queries). Flash-style streaming over keys in chunks of 32.
// qkv layout: [tok][3C], q at h*64, k at C+h*64; V comes pre-transposed (vt).
__global__ __launch_bounds__(256) void attn_kernel(const h16* __restrict__ qkv,
                                                   const h16* __restrict__ vt,
                                                   const float* __restrict__ lsz,
                                                   h16* __restrict__ outH) {
    __shared__ h16 plds[8][16 * 32];
    int lane = threadIdx.x & 31, w = threadIdx.x >> 5;
    int wid = blockIdx.x * 8 + w;     // 0..16383
    int qt = wid & 63;
    int h  = (wid >> 6) & 15;
    int b  = wid >> 10;
    const int LD = 3 * C_;
    size_t tokBase = (size_t)b * N_;
    const h16* qp  = qkv + tokBase * LD + h * HD_;
    const h16* kp  = qp + C_;
    const h16* vtp = vt + (size_t)(b * H_ + h) * HD_ * N_;   // [d][keys]
    int q0 = qt * 16;

    v16h qf0 = load_frag(qp, LD, q0 + (lane & 15), 0,  lane);
    v16h qf1 = load_frag(qp, LD, q0 + (lane & 15), 32, lane);

    v8f o[4];
#pragma unroll
    for (int j = 0; j < 4; ++j) o[j] = (v8f){};
    float rowm[8], rowl[8];
#pragma unroll
    for (int g = 0; g < 8; ++g) { rowm[g] = -1e30f; rowl[g] = 0.f; }
    const float scale = 0.125f;  // HD^-0.5

    for (int kc = 0; kc < N_; kc += 32) {
        v8f s0 = {}, s1 = {};
        {
            v16h bf0 = load_frag(kp, LD, kc + (lane & 15), 0,  lane);
            v16h bf1 = load_frag(kp, LD, kc + (lane & 15), 32, lane);
            s0 = wmma_f16(qf0, bf0, s0);
            s0 = wmma_f16(qf1, bf1, s0);
        }
        {
            v16h bf0 = load_frag(kp, LD, kc + 16 + (lane & 15), 0,  lane);
            v16h bf1 = load_frag(kp, LD, kc + 16 + (lane & 15), 32, lane);
            s1 = wmma_f16(qf0, bf0, s1);
            s1 = wmma_f16(qf1, bf1, s1);
        }
        float l0 = lsz[b * N_ + kc + (lane & 15)];
        float l1 = lsz[b * N_ + kc + 16 + (lane & 15)];
#pragma unroll
        for (int g = 0; g < 8; ++g) {
            float a = s0[g] * scale + l0;
            float c = s1[g] * scale + l1;
            float cm = redmax16(fmaxf(a, c));
            float nm = fmaxf(rowm[g], cm);
            float esc = __expf(rowm[g] - nm);
            float p0 = __expf(a - nm);
            float p1 = __expf(c - nm);
            rowl[g] = rowl[g] * esc + redsum16(p0 + p1);
            rowm[g] = nm;
#pragma unroll
            for (int j = 0; j < 4; ++j) o[j][g] *= esc;
            int r = g + ((lane >> 4) << 3);            // C-layout row
            plds[w][r * 32 + (lane & 15)]      = (h16)p0;
            plds[w][r * 32 + 16 + (lane & 15)] = (h16)p1;
        }
        // Re-shape P: C-layout (LDS) -> A-fragment layout (regs)
        v16h pa = {};
        {
            int r = lane & 15;
            int koff = (lane >> 4) << 3;
#pragma unroll
            for (int e = 0; e < 8; ++e) {
                pa[e]     = plds[w][r * 32 + koff + e];
                pa[e + 8] = plds[w][r * 32 + 16 + koff + e];
            }
        }
        // o += P(16x32) @ V(32x64): B-fragments from V^T rows (contiguous keys)
#pragma unroll
        for (int j = 0; j < 4; ++j) {
            v16h vf = load_frag(vtp, N_, 16 * j + (lane & 15), kc, lane);
            o[j] = wmma_f16(pa, vf, o[j]);
        }
    }
    int mb = (lane >> 4) << 3;
#pragma unroll
    for (int g = 0; g < 8; ++g) {
        int m = q0 + mb + g;
        float inv = 1.f / rowl[g];
        size_t rb = (tokBase + m) * C_ + h * HD_;
#pragma unroll
        for (int j = 0; j < 4; ++j)
            outH[rb + 16 * j + (lane & 15)] = (h16)(o[j][g] * inv);
    }
}

// ---------------------------------------------------------------- matching
// metric = k.mean(heads); mnorm = metric/||metric|| (f16). One wave per token.
__global__ __launch_bounds__(256) void metric_kernel(const h16* __restrict__ qkv,
                                                     h16* __restrict__ mnorm) {
    int lane = threadIdx.x & 31, w = threadIdx.x >> 5;
    int tok = blockIdx.x * 8 + w;
    const h16* kp = qkv + (size_t)tok * (3 * C_) + C_;
    float m0 = 0.f, m1 = 0.f;
#pragma unroll
    for (int h = 0; h < H_; ++h) {
        m0 += (float)kp[h * HD_ + lane];
        m1 += (float)kp[h * HD_ + 32 + lane];
    }
    m0 *= (1.f / H_); m1 *= (1.f / H_);
    float ss = m0 * m0 + m1 * m1;
    for (int m = 1; m < 32; m <<= 1) ss += __shfl_xor(ss, m, 32);
    float rn = rsqrtf(ss);
    mnorm[(size_t)tok * HD_ + lane]      = (h16)(m0 * rn);
    mnorm[(size_t)tok * HD_ + 32 + lane] = (h16)(m1 * rn);
}

// scores row max/argmax. One wave per (b, t); a = even tokens, bset = odd.
__global__ __launch_bounds__(256) void score_kernel(const h16* __restrict__ mnorm,
                                                    float* __restrict__ node_max,
                                                    int* __restrict__ node_idx) {
    __shared__ h16 aS[8][64];
    int lane = threadIdx.x & 31, w = threadIdx.x >> 5;
    int wid = blockIdx.x * 8 + w;
    int b = wid >> 9;
    int t = wid & (N2_ - 1);
    const h16* arow = mnorm + ((size_t)b * N_ + 2 * t) * HD_;
    aS[w][lane]      = arow[lane];
    aS[w][lane + 32] = arow[lane + 32];
    float bestv = -1e30f; int besti = 0;
    for (int it = 0; it < N2_ / 32; ++it) {
        int s = lane + it * 32;
        const h16* brow = mnorm + ((size_t)b * N_ + 2 * s + 1) * HD_;
        float dot = 0.f;
#pragma unroll
        for (int d = 0; d < HD_; ++d) dot += (float)aS[w][d] * (float)brow[d];
        if (dot > bestv) { bestv = dot; besti = s; }
    }
    for (int m = 1; m < 32; m <<= 1) {
        float ov = __shfl_xor(bestv, m, 32);
        int   oi = __shfl_xor(besti, m, 32);
        if (ov > bestv || (ov == bestv && oi < besti)) { bestv = ov; besti = oi; }
    }
    if (t == 0) { bestv = -1e30f; besti = 0; }   // scores[:,0,:] = -inf
    if (lane == 0) {
        node_max[b * N2_ + t] = bestv;
        node_idx[b * N2_ + t] = besti;
    }
}

// Per-batch bitonic sorts: edge_idx = argsort(-node_max) (stable via idx tie),
// then src/dst extraction and ascending sort of the unmerged 384 indices.
__global__ __launch_bounds__(512) void match_sort(const float* __restrict__ node_max,
                                                  const int* __restrict__ node_idx,
                                                  int* __restrict__ unm,
                                                  int* __restrict__ srcI,
                                                  int* __restrict__ dstI) {
    int b = blockIdx.x, t = threadIdx.x;
    __shared__ float v[N2_];
    __shared__ int   id[N2_];
    __shared__ int   u[N2_];
    v[t] = node_max[b * N2_ + t];
    id[t] = t;
    __syncthreads();
    // descending by (value desc, idx asc)
    for (int k = 2; k <= N2_; k <<= 1) {
        for (int j = k >> 1; j > 0; j >>= 1) {
            int ixj = t ^ j;
            if (ixj > t) {
                bool desc = ((t & k) == 0);
                float va = v[t], vb = v[ixj];
                int ia = id[t], ib = id[ixj];
                bool aFirst = (va > vb) || (va == vb && ia < ib);
                bool doswap = desc ? (!aFirst) : aFirst;
                if (doswap) { v[t] = vb; v[ixj] = va; id[t] = ib; id[ixj] = ia; }
            }
            __syncthreads();
        }
    }
    if (t < R_) {
        int si = id[t];
        srcI[b * R_ + t] = si;
        dstI[b * R_ + t] = node_idx[b * N2_ + si];
    }
    u[t] = (t < NU_) ? id[t + R_] : 0x7fffffff;
    __syncthreads();
    // ascending int sort of unmerged indices
    for (int k = 2; k <= N2_; k <<= 1) {
        for (int j = k >> 1; j > 0; j >>= 1) {
            int ixj = t ^ j;
            if (ixj > t) {
                bool asc = ((t & k) == 0);
                int ua = u[t], ub = u[ixj];
                bool doswap = asc ? (ua > ub) : (ua < ub);
                if (doswap) { u[t] = ub; u[ixj] = ua; }
            }
            __syncthreads();
        }
    }
    if (t < NU_) unm[b * NU_ + t] = u[t];
}

// ---------------------------------------------------------------- merge
__global__ __launch_bounds__(256) void merge_build(const float* __restrict__ x1,
                                                   const float* __restrict__ size,
                                                   const int* __restrict__ unm,
                                                   float* __restrict__ xm,
                                                   float* __restrict__ nsz) {
    int b = blockIdx.x / NM_;
    int i = blockIdx.x % NM_;
    int srcTok = (i < NU_) ? (2 * unm[b * NU_ + i]) : (2 * (i - NU_) + 1);
    float wsz = size[b * N_ + srcTok];
    const float* xr = x1 + ((size_t)b * N_ + srcTok) * C_;
    float* orow = xm + ((size_t)b * NM_ + i) * C_;
#pragma unroll
    for (int k = 0; k < 4; ++k) {
        int c = threadIdx.x + 256 * k;
        orow[c] = xr[c] * wsz;
    }
    if (threadIdx.x == 0) nsz[b * NM_ + i] = wsz;
}

__global__ __launch_bounds__(256) void merge_scatter(const float* __restrict__ x1,
                                                     const float* __restrict__ size,
                                                     const int* __restrict__ srcI,
                                                     const int* __restrict__ dstI,
                                                     float* __restrict__ xm,
                                                     float* __restrict__ nsz) {
    int b = blockIdx.x / R_;
    int s = blockIdx.x % R_;
    int srcTok = 2 * srcI[b * R_ + s];
    int row = NU_ + dstI[b * R_ + s];
    float wsz = size[b * N_ + srcTok];
    const float* xr = x1 + ((size_t)b * N_ + srcTok) * C_;
    float* orow = xm + ((size_t)b * NM_ + row) * C_;
#pragma unroll
    for (int k = 0; k < 4; ++k) {
        int c = threadIdx.x + 256 * k;
        atomicAdd(&orow[c], xr[c] * wsz);
    }
    if (threadIdx.x == 0) atomicAdd(&nsz[b * NM_ + row], wsz);
}

__global__ __launch_bounds__(256) void merge_div(float* __restrict__ xm,
                                                 const float* __restrict__ nsz) {
    int row = blockIdx.x;
    float inv = 1.f / nsz[row];
    float* orow = xm + (size_t)row * C_;
#pragma unroll
    for (int k = 0; k < 4; ++k) orow[threadIdx.x + 256 * k] *= inv;
}

// ---------------------------------------------------------------- launch
extern "C" void kernel_launch(void* const* d_in, const int* in_sizes, int n_in,
                              void* d_out, int out_size, void* d_ws, size_t ws_size,
                              hipStream_t stream) {
    const float* x       = (const float*)d_in[0];   // (B,N,C)
    const float* size    = (const float*)d_in[1];   // (B,N,1)
    const float* qkv_w   = (const float*)d_in[3];
    const float* qkv_b   = (const float*)d_in[4];
    const float* proj_w  = (const float*)d_in[5];
    const float* proj_b  = (const float*)d_in[6];
    const float* norm1_g = (const float*)d_in[7];
    const float* norm1_b = (const float*)d_in[8];
    const float* norm2_g = (const float*)d_in[9];
    const float* norm2_b = (const float*)d_in[10];
    const float* fc1_w   = (const float*)d_in[11];
    const float* fc1_b   = (const float*)d_in[12];
    const float* fc2_w   = (const float*)d_in[13];
    const float* fc2_b   = (const float*)d_in[14];
    float* out = (float*)d_out;                     // (B,NM,C) f32
    char* ws = (char*)d_ws;

    // ---- workspace layout (phase-aliased; peak ~335 MB) ----
    size_t off = 0;
    auto take = [&](size_t bytes) { size_t o = off; off += (bytes + 255) & ~(size_t)255; return o; };
    h16*   wqkvh  = (h16*)(ws + take((size_t)3 * C_ * C_ * 2));     // 6 MB
    h16*   wprojh = (h16*)(ws + take((size_t)C_ * C_ * 2));         // 2 MB
    h16*   wfc1h  = (h16*)(ws + take((size_t)C4_ * C_ * 2));        // 8 MB
    h16*   wfc2h  = (h16*)(ws + take((size_t)C_ * C4_ * 2));        // 8 MB
    size_t xh_off = take((size_t)M1_ * C_ * 2);                     // 32 MB
    h16*   xh     = (h16*)(ws + xh_off);
    size_t qkvh_off = take((size_t)M1_ * 3 * C_ * 2);               // 96 MB
    h16*   qkvh   = (h16*)(ws + qkvh_off);
    h16*   attnh  = (h16*)(ws + take((size_t)M1_ * C_ * 2));        // 32 MB
    float* x1     = (float*)(ws + take((size_t)M1_ * C_ * 4));      // 64 MB
    float* lsz    = (float*)(ws + take((size_t)B_ * N_ * 4));
    h16*   mnorm  = (h16*)(ws + take((size_t)M1_ * HD_ * 2));
    float* nmax   = (float*)(ws + take((size_t)B_ * N2_ * 4));
    int*   nidx   = (int*)(ws + take((size_t)B_ * N2_ * 4));
    int*   unm    = (int*)(ws + take((size_t)B_ * NU_ * 4));
    int*   srcI   = (int*)(ws + take((size_t)B_ * R_ * 4));
    int*   dstI   = (int*)(ws + take((size_t)B_ * R_ * 4));
    float* nsz    = (float*)(ws + take((size_t)M2_ * 4));
    float* xm     = (float*)(ws + take((size_t)M2_ * C_ * 4));      // 56 MB
    h16*   h16buf = (h16*)(ws + take((size_t)M2_ * C_ * 2));        // 28 MB
    // Aliases over dead regions:
    //  vt (32 MB) reuses xh (dead after QKV GEMM)
    //  h1 (112 MB f16) reuses qkvh+attnh (dead after proj/metric)
    h16*   vt = (h16*)(ws + xh_off);       // (B*H, 64, 1024) = V^T per head
    h16*   h1 = (h16*)(ws + qkvh_off);
    (void)ws_size; (void)in_sizes; (void)n_in; (void)out_size;

    // 1) weights -> f16
    {
        int n;
        n = 3 * C_ * C_;  cvt_f32_to_f16<<<(n + 1023) / 1024, 256, 0, stream>>>(qkv_w, wqkvh, n);
        n = C_ * C_;      cvt_f32_to_f16<<<(n + 1023) / 1024, 256, 0, stream>>>(proj_w, wprojh, n);
        n = C4_ * C_;     cvt_f32_to_f16<<<(n + 1023) / 1024, 256, 0, stream>>>(fc1_w, wfc1h, n);
        n = C_ * C4_;     cvt_f32_to_f16<<<(n + 1023) / 1024, 256, 0, stream>>>(fc2_w, wfc2h, n);
    }
    // 2) log(size)
    log_kernel<<<(B_ * N_ + 255) / 256, 256, 0, stream>>>(size, lsz, B_ * N_);
    // 3) LN1 -> f16
    ln_kernel<<<M1_, 256, 0, stream>>>(x, norm1_g, norm1_b, xh);
    // 4) QKV GEMM (16384 x 3072 x 1024)
    gemm_wmma<0, 0, 1><<<(M1_ / 64) * (3 * C_ / 64), 256, 0, stream>>>(
        xh, wqkvh, qkv_b, nullptr, nullptr, qkvh, M1_, 3 * C_, C_);
    // 5) per-head V transpose (xh is dead; vt aliases it)
    vtrans_kernel<<<B_ * H_ * (N_ / 64), 256, 0, stream>>>(qkvh, vt);
    // 6) flash attention (16 q / wave), V^T fragments are 16B vector loads
    attn_kernel<<<M1_ / 8, 256, 0, stream>>>(qkvh, vt, lsz, attnh);
    // 7) metric = mean-over-heads(k), normalized
    metric_kernel<<<M1_ / 8, 256, 0, stream>>>(qkvh, mnorm);
    // 8) proj GEMM + residual x -> x1 (f32)
    gemm_wmma<0, 1, 0><<<(M1_ / 64) * (C_ / 64), 256, 0, stream>>>(
        attnh, wprojh, proj_b, x, x1, nullptr, M1_, C_, C_);
    // 9) bipartite scores: per-row max/argmax
    score_kernel<<<(B_ * N2_) / 8, 256, 0, stream>>>(mnorm, nmax, nidx);
    // 10) per-batch sorts -> unm/src/dst
    match_sort<<<B_, 512, 0, stream>>>(nmax, nidx, unm, srcI, dstI);
    // 11) merge: gather (x1*size), scatter-add, divide by new_size
    merge_build<<<B_ * NM_, 256, 0, stream>>>(x1, size, unm, xm, nsz);
    merge_scatter<<<B_ * R_, 256, 0, stream>>>(x1, size, srcI, dstI, xm, nsz);
    merge_div<<<M2_, 256, 0, stream>>>(xm, nsz);
    // 12) LN2 -> f16
    ln_kernel<<<M2_, 256, 0, stream>>>(xm, norm2_g, norm2_b, h16buf);
    // 13) FC1 + exact GELU (14336 x 4096 x 1024)
    gemm_wmma<1, 0, 1><<<(M2_ / 64) * (C4_ / 64), 256, 0, stream>>>(
        h16buf, wfc1h, fc1_b, nullptr, nullptr, h1, M2_, C4_, C_);
    // 14) FC2 + residual xm -> d_out (14336 x 1024 x 4096)
    gemm_wmma<0, 1, 0><<<(M2_ / 64) * (C_ / 64), 256, 0, stream>>>(
        h1, wfc2h, fc2_b, xm, out, nullptr, M2_, C_, C4_);
}
